// RefLocal_85375359910440
// MI455X (gfx1250) — matrix operand order repaired and checked by reading, not compiled, not run on hardware
//
#include <hip/hip_runtime.h>

typedef __attribute__((ext_vector_type(16))) _Float16 v16h;
typedef __attribute__((ext_vector_type(8)))  _Float16 v8h;
typedef __attribute__((ext_vector_type(8)))  float    v8f;

// Problem constants (reference: B=2, H=W=128, C=BIN=64, k=5)
constexpr int kH   = 128;
constexpr int kW   = 128;
constexpr int kC   = 64;
constexpr int kBIN = 64;
constexpr int kKS  = 5;
constexpr int kPAD = 2;
constexpr int TILE = 16;          // pixels per wave (one row segment)
constexpr int HP   = kH + 4;      // padded height (132)
constexpr int WP   = 144;         // padded width: W+4 rounded up so tap over-reads hit zeros

// ---------------------------------------------------------------------------
// Pre-pass 1: ref (f32, [B,H,W,C]) -> refh (f16, [B,HP,WP,C]) zero-padded.
__global__ __launch_bounds__(256)
void pack_ref(const float* __restrict__ src, _Float16* __restrict__ dst, int B)
{
    size_t idx = (size_t)blockIdx.x * 256 + threadIdx.x;
    size_t total = (size_t)B * HP * WP * (kC / 8);
    if (idx >= total) return;
    int c8 = (int)(idx % (kC / 8));
    size_t t = idx / (kC / 8);
    int wp = (int)(t % WP); t /= WP;
    int hp = (int)(t % HP); t /= HP;
    int b  = (int)t;
    int h = hp - kPAD, w = wp - kPAD;
    float f[8];
    if (h >= 0 && h < kH && w >= 0 && w < kW) {
        const float* p = src + (((size_t)b * kH + h) * kW + w) * kC + c8 * 8;
        float4 f0 = *(const float4*)p;
        float4 f1 = *(const float4*)(p + 4);
        f[0]=f0.x; f[1]=f0.y; f[2]=f0.z; f[3]=f0.w;
        f[4]=f1.x; f[5]=f1.y; f[6]=f1.z; f[7]=f1.w;
    } else {
        #pragma unroll
        for (int j = 0; j < 8; ++j) f[j] = 0.0f;
    }
    v8h o;
    #pragma unroll
    for (int j = 0; j < 8; ++j) o[j] = (_Float16)f[j];
    *(v8h*)(dst + idx * 8) = o;
}

// Pre-pass 2: plain f32 -> f16 convert (main), 8 elements per thread.
__global__ __launch_bounds__(256)
void cvt_f16(const float* __restrict__ src, _Float16* __restrict__ dst, size_t n8)
{
    size_t idx = (size_t)blockIdx.x * 256 + threadIdx.x;
    if (idx >= n8) return;
    const float* p = src + idx * 8;
    float4 f0 = *(const float4*)p;
    float4 f1 = *(const float4*)(p + 4);
    v8h o;
    o[0]=(_Float16)f0.x; o[1]=(_Float16)f0.y; o[2]=(_Float16)f0.z; o[3]=(_Float16)f0.w;
    o[4]=(_Float16)f1.x; o[5]=(_Float16)f1.y; o[6]=(_Float16)f1.z; o[7]=(_Float16)f1.w;
    *(v8h*)(dst + idx * 8) = o;
}

// Pre-pass 3: ref_value (f32, [B,H,W,BIN]) -> valT (f16, [B,HP,BIN,WP]),
// transposed (spatial innermost) + zero-padded.
__global__ __launch_bounds__(256)
void pack_valT(const float* __restrict__ src, _Float16* __restrict__ dst, int B)
{
    size_t idx = (size_t)blockIdx.x * 256 + threadIdx.x;
    size_t total = (size_t)B * HP * kBIN * (WP / 8);
    if (idx >= total) return;
    int wp8 = (int)(idx % (WP / 8));
    size_t t = idx / (WP / 8);
    int bin = (int)(t % kBIN); t /= kBIN;
    int hp  = (int)(t % HP);   t /= HP;
    int b   = (int)t;
    int h = hp - kPAD;
    bool hok = (h >= 0 && h < kH);
    v8h o;
    #pragma unroll
    for (int j = 0; j < 8; ++j) {
        int w = wp8 * 8 + j - kPAD;
        float f = 0.0f;
        if (hok && w >= 0 && w < kW)
            f = src[(((size_t)b * kH + h) * kW + w) * kBIN + bin];
        o[j] = (_Float16)f;
    }
    *(v8h*)(dst + idx * 8) = o;
}

// ---------------------------------------------------------------------------
// Hot kernel: one wave32 per 16-pixel row tile.
__global__ __launch_bounds__(32)
void local_attn_wmma(const _Float16* __restrict__ mainh,
                     const _Float16* __restrict__ refh,
                     const _Float16* __restrict__ valT,
                     float* __restrict__ outp)
{
    __shared__ float    G[kKS][TILE][32];  // raw scores G[di][pixel][ref-col 0..19]
    __shared__ _Float16 BND[TILE][32];     // banded attention row per pixel (f16)

    const int lane = threadIdx.x;
    const int g    = lane >> 4;            // half-wave group (K-group of fragments)
    const int ln   = lane & 15;

    const int tile = blockIdx.x;
    const int tw   = tile % (kW / TILE);
    const int th   = (tile / (kW / TILE)) % kH;
    const int b    = tile / ((kW / TILE) * kH);
    const int w0   = tw * TILE;

    // Zero the band rows once: lane (ln,g) owns BND[ln][16g..16g+15].
    {
        v16h z = {};
        *(v16h*)&BND[ln][16 * g] = z;
    }

    // A fragment (phase A): main tile 16x64, two K=32 chunks.
    // f16 A 16x32 layout: lane M=ln, element runs c = 8g+32q+[0,8) and 16+8g+32q+[0,8)
    v16h amain[2];
    {
        const _Float16* mrow = mainh + (((size_t)b * kH + th) * kW + (w0 + ln)) * kC;
        #pragma unroll
        for (int q = 0; q < 2; ++q) {
            v8h lo = *(const v8h*)(mrow + 8 * g + 32 * q);
            v8h hi = *(const v8h*)(mrow + 16 + 8 * g + 32 * q);
            amain[q] = __builtin_shufflevector(lo, hi, 0,1,2,3,4,5,6,7,8,9,10,11,12,13,14,15);
        }
    }

    // -------- Phase A: G[di] = M(16x64) x R^T(64x20 pad 32) ----------------
    #pragma unroll
    for (int di = 0; di < kKS; ++di) {
        const size_t rowbase = (((size_t)b * HP + (th + di)) * WP);
        #pragma unroll
        for (int nt = 0; nt < 2; ++nt) {
            v8f acc = {};
            const size_t pix = rowbase + (w0 + ln + 16 * nt);   // padded col index
            #pragma unroll
            for (int q = 0; q < 2; ++q) {
                // B fragment: lane N=ln, c runs 8g+32q+[0,8) and 16+8g+32q+[0,8) (matches A)
                v8h blo = *(const v8h*)(refh + pix * kC + 8 * g + 32 * q);
                v8h bhi = *(const v8h*)(refh + pix * kC + 16 + 8 * g + 32 * q);
                v16h bf = __builtin_shufflevector(blo, bhi, 0,1,2,3,4,5,6,7,8,9,10,11,12,13,14,15);
                acc = __builtin_amdgcn_wmma_f32_16x16x32_f16(
                          false, amain[q], false, bf, (short)0, acc, false, false);
            }
            #pragma unroll
            for (int r = 0; r < 8; ++r)
                G[di][r + 8 * g][ln + 16 * nt] = acc[r];
        }
    }
    __syncthreads();

    // -------- Softmax over 25 taps per pixel (in registers) ----------------
    float sm[25];
    float sminv;
    {
        const int i = ln;                  // both half-waves compute pixel ln
        float mx = -3.0e38f;
        #pragma unroll
        for (int p = 0; p < 25; ++p) {
            sm[p] = G[p / 5][i][i + (p % 5)];   // zero-padded taps score 0 (matches ref)
            mx = fmaxf(mx, sm[p]);
        }
        float sum = 0.0f;
        #pragma unroll
        for (int p = 0; p < 25; ++p) { sm[p] = __expf(sm[p] - mx); sum += sm[p]; }
        sminv = 1.0f / sum;
    }

    // -------- Phase B (swapped operands): D(bins x pixels) -----------------
    //   A = V^T : M=bin, K=spatial taps     (two 16B runs from valT)
    //   B = band: K=spatial taps, N=pixel   (two 16B LDS reads per lane)
    v8f oacc[4] = {v8f{}, v8f{}, v8f{}, v8f{}};
    #pragma unroll
    for (int di = 0; di < kKS; ++di) {
        // Scatter this di's 5 attention taps into the band row: BND[i][i+d].
        if (g == 0) {
            #pragma unroll
            for (int d = 0; d < 5; ++d)
                BND[ln][ln + d] = (_Float16)(sm[5 * di + d] * sminv);
        }
        __syncthreads();
        // B fragment: lane N = pixel ln, k runs 8g+[0,8) and 16+8g+[0,8) (matches A)
        v8h blo = *(const v8h*)&BND[ln][8 * g];
        v8h bhi = *(const v8h*)&BND[ln][16 + 8 * g];
        v16h bf = __builtin_shufflevector(blo, bhi, 0,1,2,3,4,5,6,7,8,9,10,11,12,13,14,15);

        const size_t vrow = (((size_t)b * HP + (th + di)) * kBIN);
        #pragma unroll
        for (int t = 0; t < 4; ++t) {
            // A fragment: lane M = bin ln+16t, K runs w0+8g+[0,8) and w0+16+8g+[0,8)
            const _Float16* vp = valT + (vrow + (ln + 16 * t)) * WP + w0;
            v8h lo = *(const v8h*)(vp + 8 * g);
            v8h hi = *(const v8h*)(vp + 16 + 8 * g);
            v16h af = __builtin_shufflevector(lo, hi, 0,1,2,3,4,5,6,7,8,9,10,11,12,13,14,15);
            oacc[t] = __builtin_amdgcn_wmma_f32_16x16x32_f16(
                          false, af, false, bf, (short)0, oacc[t], false, false);
        }
        __syncthreads();
    }

    // -------- Store D tiles: lane = pixel, VGPR r -> bin 16t+8g+r ----------
    {
        float* orow = outp + (((size_t)b * kH + th) * kW + (w0 + ln)) * kBIN;
        #pragma unroll
        for (int t = 0; t < 4; ++t)
            *(v8f*)(orow + 16 * t + 8 * g) = oacc[t];   // 32B aligned vector store
    }
}

// ---------------------------------------------------------------------------
extern "C" void kernel_launch(void* const* d_in, const int* in_sizes, int n_in,
                              void* d_out, int out_size, void* d_ws, size_t ws_size,
                              hipStream_t stream) {
    const float* mainp = (const float*)d_in[0];
    const float* refp  = (const float*)d_in[1];
    const float* valp  = (const float*)d_in[2];
    float* outp        = (float*)d_out;

    const int B = in_sizes[0] / (kH * kW * kC);   // = 2

    // Workspace layout (all 16B-aligned)
    const size_t nRef  = (size_t)B * HP * WP * kC;     // f16 elems
    const size_t nMain = (size_t)B * kH * kW * kC;
    char* ws = (char*)d_ws;
    _Float16* refh  = (_Float16*)ws;
    _Float16* mainh = (_Float16*)(ws + nRef * sizeof(_Float16));
    _Float16* valT  = (_Float16*)(ws + (nRef + nMain) * sizeof(_Float16));

    // Pre-pass: pad + convert + transpose
    {
        size_t t1 = (size_t)B * HP * WP * (kC / 8);
        pack_ref<<<(unsigned)((t1 + 255) / 256), 256, 0, stream>>>(refp, refh, B);
        size_t t2 = nMain / 8;
        cvt_f16<<<(unsigned)((t2 + 255) / 256), 256, 0, stream>>>(mainp, mainh, t2);
        size_t t3 = (size_t)B * HP * kBIN * (WP / 8);
        pack_valT<<<(unsigned)((t3 + 255) / 256), 256, 0, stream>>>(valp, valT, B);
    }

    // Hot kernel: one wave32 per 16-pixel tile
    const int tiles = B * kH * (kW / TILE);
    local_attn_wmma<<<tiles, 32, 0, stream>>>(mainh, refh, valT, outp);
}
